// SlayerMM_841813590370
// MI455X (gfx1250) — matrix operand order repaired and checked by reading, not compiled, not run on hardware
//
#include <hip/hip_runtime.h>

typedef __attribute__((ext_vector_type(16))) _Float16 v16h;
typedef __attribute__((ext_vector_type(8)))  float    v8f;

#define THETA      10.0f
#define DECAY_SR   0.9048374180359595f   // exp(-1/10)
#define DECAY_REF  0.36787944117144233f  // exp(-1)
#define REF_KICK   20.0f                 // SCALE_REF * THETA

static constexpr int kT = 500;           // timesteps (fixed by reference)

union HBits { _Float16 h; unsigned short u; };

// ---------------------------------------------------------------------------
// Weight convert: f32 (O,F) -> f16 (O,Fp) with zero padding (Fp = ceil32(F)).
// Zero padding keeps OOB-k A fragments exact (no NaN-bit garbage).
// ---------------------------------------------------------------------------
__global__ void cvt_w_pad(const float* __restrict__ src, _Float16* __restrict__ dst,
                          int O, int F, int Fp)
{
    const long long idx = (long long)blockIdx.x * blockDim.x + threadIdx.x;
    if (idx >= (long long)O * Fp) return;
    const int r = (int)(idx / Fp);
    const int c = (int)(idx - (long long)r * Fp);
    const float v = (c < F) ? src[(size_t)r * F + c] : 0.0f;
    dst[idx] = (_Float16)v;
}

// ---------------------------------------------------------------------------
// Activation convert: f32 (NB,F,T) -> packed f16 words (NB, F/2, T):
//   word[(n*F/2 + fp)*T + t] = { f16(x[2fp][t]), f16(x[2fp+1][t]) }
// Adjacent-k pairs pre-packed => B staging needs zero VALU packing.
// ---------------------------------------------------------------------------
__global__ void cvt_x_pack(const float* __restrict__ src, unsigned int* __restrict__ dst,
                           long long total)
{
    const long long w = (long long)blockIdx.x * blockDim.x + threadIdx.x;
    if (w >= total) return;
    const long long row = w / kT;           // = n*(F/2) + fp
    const int t = (int)(w - row * kT);
    const float lo = src[(size_t)(2 * row) * kT + t];
    const float hi = src[(size_t)(2 * row + 1) * kT + t];
    HBits L, H; L.h = (_Float16)lo; H.h = (_Float16)hi;
    dst[w] = (unsigned int)L.u | ((unsigned int)H.u << 16);
}

// ---------------------------------------------------------------------------
// B staging: tile 128(t) x 32(k) halves. Each thread handles one t column and
// 8 k-pair words (coalesced b32 loads, immediate row offsets), commits 16
// contiguous halves -> 2x ds_store_b128.  OOB t clamped (cols never stored);
// OOB k zeroed (would pollute valid outputs).
// ---------------------------------------------------------------------------
__device__ __forceinline__ void fetchB(const unsigned int* __restrict__ Xw,
                                       int F, int nb, int f0, int t0,
                                       int tid, unsigned int rb[8])
{
    const int tcol = tid & 127;
    const int kg   = (tid >> 7) & 1;
    const int fb   = f0 + kg * 16;
    const int tt   = t0 + tcol;
    const int ttc  = (tt < kT) ? tt : (kT - 1);
    const int Fh   = F >> 1;
    const int fpb  = fb >> 1;
    if (fb + 16 <= F) {
        const unsigned int* xp = Xw + ((size_t)nb * Fh + fpb) * kT + ttc;
        #pragma unroll
        for (int i = 0; i < 8; ++i)
            rb[i] = xp[(size_t)i * kT];
    } else {
        #pragma unroll
        for (int i = 0; i < 8; ++i) {
            const int fp = fpb + i;
            const bool ok = (2 * fp) < F;               // F even: pairs don't straddle
            const int fpc = ok ? fp : (Fh - 1);
            const unsigned int v = Xw[((size_t)nb * Fh + fpc) * kT + ttc];
            rb[i] = ok ? v : 0u;
        }
    }
}

__device__ __forceinline__ void commitB(_Float16* __restrict__ lB,
                                        int tid, const unsigned int rb[8])
{
    const int tcol = tid & 127;
    const int kg   = (tid >> 7) & 1;
    uint4* p = (uint4*)(lB + tcol * 32 + kg * 16);
    p[0] = make_uint4(rb[0], rb[1], rb[2], rb[3]);
    p[1] = make_uint4(rb[4], rb[5], rb[6], rb[7]);
}

// ---------------------------------------------------------------------------
// WMMA GEMM: Z[n,o,t] = sum_f W[o,f] * X[n,f,t]
// Wh: f16 (O,Fp) zero-padded; Xw: packed f16 words (NB,F/2,T); Z: f32.
// 256 threads = 8 waves (4M x 2N); wave tile 32x64 -> 8 accumulators
// (2 A-frags x 4 B-frags). A fragments load DIRECTLY from global (b128,
// documented layout), only B goes through double-buffered LDS.
// ---------------------------------------------------------------------------
__launch_bounds__(256)
__global__ void slayer_gemm_wmma(const _Float16* __restrict__ Wh,
                                 const unsigned int* __restrict__ Xw,
                                 float* __restrict__ Z,
                                 int O, int F, int Fp)
{
    __shared__ __align__(16) _Float16 ldsB[2][128 * 32];  // [buf][t][k]

    const int tid  = threadIdx.x;
    const int nb   = blockIdx.z;
    const int o0   = blockIdx.y * 128;
    const int t0   = blockIdx.x * 128;

    const int wave = tid >> 5;
    const int lane = tid & 31;
    const int mw   = wave & 3;           // 0..3 : M sub-tile (32 rows)
    const int nw   = wave >> 2;          // 0..1 : N sub-tile (64 cols)
    const int h    = lane >> 4;
    const int nlo  = lane & 15;

    union Frag { v16h h; uint4 q[2]; };

    v8f acc[2][4];
    #pragma unroll
    for (int i = 0; i < 2; ++i)
        #pragma unroll
        for (int j = 0; j < 4; ++j)
            acc[i][j] = (v8f){};

    // A rows for this lane's two fragments (clamped: garbage rows never stored,
    // and clamped rows hold real finite weights).
    const int ar0 = o0 + mw * 32 + nlo;
    const int ar1 = ar0 + 16;
    const _Float16* aRow0 = Wh + (size_t)((ar0 < O) ? ar0 : (O - 1)) * Fp;
    const _Float16* aRow1 = Wh + (size_t)((ar1 < O) ? ar1 : (O - 1)) * Fp;

    const int nChunks = (F + 31) / 32;

    unsigned int rb[8];
    fetchB(Xw, F, nb, 0, t0, tid, rb);
    commitB(ldsB[0], tid, rb);

    int cur = 0;
    for (int ic = 0; ic < nChunks; ++ic) {
        __syncthreads();

        const bool more = (ic + 1) < nChunks;
        if (more)
            fetchB(Xw, F, nb, (ic + 1) * 32, t0, tid, rb);

        const int f0 = ic * 32;

        // ---- A fragments direct from global (zero-padded rows) ------------
        Frag a[2];
        a[0].q[0] = *(const uint4*)(aRow0 + f0 + 8 * h);
        a[0].q[1] = *(const uint4*)(aRow0 + f0 + 16 + 8 * h);
        a[1].q[0] = *(const uint4*)(aRow1 + f0 + 8 * h);
        a[1].q[1] = *(const uint4*)(aRow1 + f0 + 16 + 8 * h);

        // ---- B fragments from LDS ----------------------------------------
        const _Float16* lB = ldsB[cur];
        Frag b[4];
        #pragma unroll
        for (int bj = 0; bj < 4; ++bj) {
            const int rowB = nw * 64 + bj * 16 + nlo;
            b[bj].q[0] = *(const uint4*)&lB[rowB * 32 + 8 * h];
            b[bj].q[1] = *(const uint4*)&lB[rowB * 32 + 16 + 8 * h];
        }

        #pragma unroll
        for (int ai = 0; ai < 2; ++ai)
            #pragma unroll
            for (int bj = 0; bj < 4; ++bj)
                acc[ai][bj] = __builtin_amdgcn_wmma_f32_16x16x32_f16(
                    false, a[ai].h, false, b[bj].h, (short)0,
                    acc[ai][bj], false, false);

        if (more)
            commitB(ldsB[cur ^ 1], tid, rb);
        cur ^= 1;
    }

    // ---- epilogue: D layout -> M = j + 8*h, N = lane&15 ---------------------
    union { v8f v; float f[8]; } r;
    #pragma unroll
    for (int ai = 0; ai < 2; ++ai) {
        #pragma unroll
        for (int bj = 0; bj < 4; ++bj) {
            r.v = acc[ai][bj];
            const int tcol = t0 + nw * 64 + bj * 16 + nlo;
            if (tcol < kT) {
                #pragma unroll
                for (int j = 0; j < 8; ++j) {
                    const int o = o0 + mw * 32 + ai * 16 + 8 * h + j;
                    if (o < O)
                        Z[((size_t)nb * O + o) * kT + tcol] = r.f[j];
                }
            }
        }
    }
}

// ---------------------------------------------------------------------------
// Fused psp + spike scan.  Spikes are exactly 0x0000 / 0x3C00 in f16, written
// straight into the packed (f-pair, t) activation layout -- no conversions.
// ---------------------------------------------------------------------------
__global__ void slayer_psp_spike_h(const float* __restrict__ z,
                                   unsigned short* __restrict__ s,
                                   int O, int NB, int outC, int chanOff)
{
    const int r = blockIdx.x * blockDim.x + threadIdx.x;
    if (r >= NB * O) return;
    const int n = r / O;
    const int o = r - n * O;
    const int c = chanOff + o;

    const float* zr = z + (size_t)r * kT;
    unsigned short* sr = s + (((size_t)n * (outC >> 1) + (c >> 1)) * kT) * 2 + (c & 1);

    float p = 0.0f, ref = 0.0f;
    for (int t = 0; t < kT; t += 4) {
        const float4 zq = *(const float4*)(zr + t);
        float zv[4] = { zq.x, zq.y, zq.z, zq.w };
        #pragma unroll
        for (int i = 0; i < 4; ++i) {
            p = DECAY_SR * p + zv[i];
            const bool sp = (p + ref) >= THETA;
            ref = DECAY_REF * (ref - (sp ? REF_KICK : 0.0f));
            sr[2 * (t + i)] = sp ? 0x3C00u : 0u;   // f16 1.0 / 0.0
        }
    }
}

__global__ void slayer_psp_spike_f(const float* __restrict__ z,
                                   float* __restrict__ s,
                                   int O, int NB)
{
    const int r = blockIdx.x * blockDim.x + threadIdx.x;
    if (r >= NB * O) return;

    const float* zr = z + (size_t)r * kT;
    float* sr = s + (size_t)r * kT;

    float p = 0.0f, ref = 0.0f;
    for (int t = 0; t < kT; t += 4) {
        const float4 zq = *(const float4*)(zr + t);
        float zv[4] = { zq.x, zq.y, zq.z, zq.w };
        float sv[4];
        #pragma unroll
        for (int i = 0; i < 4; ++i) {
            p = DECAY_SR * p + zv[i];
            const bool sp = (p + ref) >= THETA;
            ref = DECAY_REF * (ref - (sp ? REF_KICK : 0.0f));
            sv[i] = sp ? 1.0f : 0.0f;
        }
        *(float4*)(sr + t) = make_float4(sv[0], sv[1], sv[2], sv[3]);
    }
}

// ---------------------------------------------------------------------------
// Orchestration. N=16, T=500; tact 156->50; vis 6300->1024->10; fuse 60->20.
// psp/dense commute, so every stage = binary-spike GEMM -> psp -> spike.
// ---------------------------------------------------------------------------
extern "C" void kernel_launch(void* const* d_in, const int* in_sizes, int n_in,
                              void* d_out, int out_size, void* d_ws, size_t ws_size,
                              hipStream_t stream)
{
    (void)in_sizes; (void)n_in; (void)out_size; (void)ws_size;

    const float* xt  = (const float*)d_in[0];   // (16, 156, 500)
    const float* xv  = (const float*)d_in[1];   // (16, 6300, 500)
    const float* Wt  = (const float*)d_in[2];   // (50, 156)
    const float* Wv1 = (const float*)d_in[3];   // (1024, 6300)
    const float* Wv2 = (const float*)d_in[4];   // (10, 1024)
    const float* Wf  = (const float*)d_in[5];   // (20, 60)
    float* out = (float*)d_out;                 // (16, 20, 500)

    const int NB = 16;

    // ---- workspace layout (bytes) ----
    char* w = (char*)d_ws;
    float* zBuf = (float*)w;             w += (size_t)NB * 1024 * kT * 4;   // 32.8 MB
    unsigned int* xt16 = (unsigned int*)w;  w += (size_t)NB * 78   * kT * 4; // 2.5 MB
    unsigned int* xv16 = (unsigned int*)w;  w += (size_t)NB * 3150 * kT * 4; // 100.8 MB
    unsigned int* sv116 = (unsigned int*)w; w += (size_t)NB * 512  * kT * 4; // 16.4 MB
    unsigned int* sall16 = (unsigned int*)w; w += (size_t)NB * 30  * kT * 4; // 1.0 MB
    _Float16* Wt16  = (_Float16*)w;      w += (size_t)50   * 160  * 2;
    _Float16* Wv116 = (_Float16*)w;      w += (size_t)1024 * 6304 * 2;       // 12.9 MB
    _Float16* Wv216 = (_Float16*)w;      w += (size_t)10   * 1024 * 2;
    _Float16* Wf16  = (_Float16*)w;      w += (size_t)20   * 64   * 2;

    dim3 blk(256);
    auto cdiv = [](long long a, long long b) { return (unsigned)((a + b - 1) / b); };

    // ---- one-time precision conversion --------------------------------------
    cvt_w_pad<<<cdiv((long long)50 * 160, 256), blk, 0, stream>>>(Wt, Wt16, 50, 156, 160);
    cvt_w_pad<<<cdiv((long long)1024 * 6304, 256), blk, 0, stream>>>(Wv1, Wv116, 1024, 6300, 6304);
    cvt_w_pad<<<cdiv((long long)10 * 1024, 256), blk, 0, stream>>>(Wv2, Wv216, 10, 1024, 1024);
    cvt_w_pad<<<cdiv((long long)20 * 64, 256), blk, 0, stream>>>(Wf, Wf16, 20, 60, 64);

    const long long nxt = (long long)NB * 78 * kT;
    const long long nxv = (long long)NB * 3150 * kT;
    cvt_x_pack<<<cdiv(nxt, 256), blk, 0, stream>>>(xt, xt16, nxt);
    cvt_x_pack<<<cdiv(nxv, 256), blk, 0, stream>>>(xv, xv16, nxv);

    const int tTiles = (kT + 127) / 128;   // 4

    // -------- stage 1: tactile (O=50, F=156) --------
    slayer_gemm_wmma<<<dim3(tTiles, 1, NB), blk, 0, stream>>>(Wt16, xt16, zBuf, 50, 156, 160);
    slayer_psp_spike_h<<<cdiv(NB * 50, 256), blk, 0, stream>>>(
        zBuf, (unsigned short*)sall16, 50, NB, 60, 0);

    // -------- stage 2: vision hidden (O=1024, F=6300) — dominant GEMM --------
    slayer_gemm_wmma<<<dim3(tTiles, 8, NB), blk, 0, stream>>>(Wv116, xv16, zBuf, 1024, 6300, 6304);
    slayer_psp_spike_h<<<cdiv(NB * 1024, 256), blk, 0, stream>>>(
        zBuf, (unsigned short*)sv116, 1024, NB, 1024, 0);

    // -------- stage 3: vision out (O=10, F=1024) --------
    slayer_gemm_wmma<<<dim3(tTiles, 1, NB), blk, 0, stream>>>(Wv216, sv116, zBuf, 10, 1024, 1024);
    slayer_psp_spike_h<<<cdiv(NB * 10, 256), blk, 0, stream>>>(
        zBuf, (unsigned short*)sall16, 10, NB, 60, 50);

    // -------- stage 4: fusion classifier (O=20, F=60) --------
    slayer_gemm_wmma<<<dim3(tTiles, 1, NB), blk, 0, stream>>>(Wf16, sall16, zBuf, 20, 60, 64);
    slayer_psp_spike_f<<<cdiv(NB * 20, 256), blk, 0, stream>>>(zBuf, out, 20, NB);
}
